// QuantizedLoRALinear_83451214561340
// MI455X (gfx1250) — compile-verified, matrix operand order checked by reading, not run on hardware
//
#include <hip/hip_runtime.h>

#define IN_F   8192
#define OUT_F  8192
#define M_TOK  64
#define NT     128           // output columns per workgroup
#define KT     128           // K depth staged in LDS per iteration
#define KSPLIT 4
#define KCHUNK (IN_F / KSPLIT)   // 2048
#define LDSS   136           // padded LDS row stride in halves (272B, 16B-aligned, bank-rotating)

#if __has_builtin(__builtin_amdgcn_global_load_async_to_lds_b128)
#define HAVE_ASYNC_LDS 1
#else
#define HAVE_ASYNC_LDS 0
#endif

typedef __attribute__((ext_vector_type(16))) _Float16 v16h;
typedef __attribute__((ext_vector_type(8)))  _Float16 v8h;
typedef __attribute__((ext_vector_type(4)))  _Float16 v4h;
typedef __attribute__((ext_vector_type(2)))  _Float16 v2h;
typedef __attribute__((ext_vector_type(8)))  float    v8f;
typedef __attribute__((ext_vector_type(4)))  int      v4i;

// address-space-qualified v4i for the async-to-LDS builtin
typedef __attribute__((address_space(1))) v4i av1_v4i;   // global
typedef __attribute__((address_space(3))) v4i av3_v4i;   // LDS

__device__ __forceinline__ void wait_async0() {
#if __has_builtin(__builtin_amdgcn_s_wait_asynccnt)
    __builtin_amdgcn_s_wait_asynccnt(0);
#else
    asm volatile("s_wait_asynccnt 0x0" ::: "memory");
#endif
}

// ---------------------------------------------------------------------------
// Packed-magic 4-bit dequant: byte b -> {(qlo-8)*s, (qhi-8)*s} as f16 pair.
//   d = (b&0xF) | ((b<<12)&0xF0000) | 0x64006400  ==  {1024+qlo, 1024+qhi} f16
//   (d + {-1032,-1032}) is the EXACT integer q-8, then one rounding by *s.
// Lowers to ~3 bit-ops + v_pk_add_f16 + v_pk_mul_f16 per 2 elements.
// ---------------------------------------------------------------------------
__device__ __forceinline__ v2h dq2(int b, v2h s2) {
    unsigned d = ((unsigned)b & 15u) | (((unsigned)b << 12) & 0x000F0000u) | 0x64006400u;
    v2h q = __builtin_bit_cast(v2h, d);
    const v2h m1032 = __builtin_bit_cast(v2h, 0xE408E408u);  // {-1032.0h, -1032.0h}
    return (q + m1032) * s2;
}

// ---------------------------------------------------------------------------
// Load one 16x32 f16 fragment (A- or B-operand) from an LDS tile stored
// row-major (row = M or N, contiguous K). Per CDNA5 ISA 16-bit A layout:
//   lanes 0-15 : row = lane,    halves 0-7 = K+0..7,  halves 8-15 = K+16..23
//   lanes 16-31: row = lane-16, halves 0-7 = K+8..15, halves 8-15 = K+24..31
// Two ds_load_b128 per fragment.
// ---------------------------------------------------------------------------
__device__ __forceinline__ v16h ld_frag(const _Float16* row, int kbase, int lane) {
    const _Float16* p = row + kbase + ((lane >> 4) << 3);
    v8h lo = *(const v8h*)(p);
    v8h hi = *(const v8h*)(p + 16);
    return __builtin_shufflevector(lo, hi, 0,1,2,3,4,5,6,7,8,9,10,11,12,13,14,15);
}

// ---------------------------------------------------------------------------
// x (f32) -> f16 workspace, 4 elements/thread
// ---------------------------------------------------------------------------
__global__ void __launch_bounds__(256) cvt_kernel(const float* __restrict__ x,
                                                  _Float16* __restrict__ x16) {
    int i = (blockIdx.x * 256 + threadIdx.x) * 4;
    float4 v = *(const float4*)(x + i);
    v4h h = { (_Float16)v.x, (_Float16)v.y, (_Float16)v.z, (_Float16)v.w };
    *(v4h*)(x16 + i) = h;
}

// ---------------------------------------------------------------------------
// xa[s][r] = dot(x[s,:], lora_A[r,:]) ; one block per (s,r)
// ---------------------------------------------------------------------------
__global__ void __launch_bounds__(256) xa_kernel(const float* __restrict__ x,
                                                 const float* __restrict__ loraA,
                                                 float* __restrict__ xa) {
    int b = blockIdx.x;
    int s = b >> 4;
    int r = b & 15;
    const float* xr = x + (size_t)s * IN_F;
    const float* ar = loraA + (size_t)r * IN_F;
    float p = 0.f;
    for (int k = threadIdx.x; k < IN_F; k += 256) p += xr[k] * ar[k];
    __shared__ float red[256];
    red[threadIdx.x] = p;
    __syncthreads();
    for (int off = 128; off > 0; off >>= 1) {
        if (threadIdx.x < off) red[threadIdx.x] += red[threadIdx.x + off];
        __syncthreads();
    }
    if (threadIdx.x == 0) xa[s * 16 + r] = red[0];
}

// ---------------------------------------------------------------------------
// Main tiled GEMM: 64 x 128 output tile per WG, K split KSPLIT ways.
// 8 waves in a 2(M) x 4(N) grid; each wave owns a 32x32 register tile
// (4 accumulators, a0/a1 x b0/b1 fragment reuse -> 2 ds_loads per WMMA).
// Software pipeline: next stage's packed weights prefetched into registers
// during the 16 WMMAs; x-tile streams via GLOBAL_LOAD_ASYNC_TO_LDS.
// ---------------------------------------------------------------------------
__global__ void __launch_bounds__(256) qlora_gemm_kernel(
        const int*   __restrict__ packed,   // NUMEL/2 int32, one byte each
        const float* __restrict__ scales,   // NUMEL/64
        const float* __restrict__ loraB,    // [OUT_F,16]
        const _Float16* __restrict__ x16,   // [64,IN_F]
        const float* __restrict__ xa,       // [64,16]
        float*       __restrict__ part)     // [KSPLIT, 64, OUT_F]
{
    __shared__ _Float16 tA[M_TOK][LDSS];   // x tile   (row = token,  cols = K)
    __shared__ _Float16 tB[NT][LDSS];      // W tile   (row = out-col, cols = K)
    __shared__ float    xs[M_TOK][16];     // LoRA xa cache

    const int tid  = threadIdx.x;
    const int lane = tid & 31;
    const int wid  = tid >> 5;
    const int nl   = lane & 15;
    const int wave_m = wid >> 2;           // 0..1
    const int wave_n = wid & 3;            // 0..3
    const int m0 = wave_m * 32;
    const int n0 = wave_n * 32;
    const int ntile = blockIdx.x * NT;
    const int kh    = blockIdx.y;          // K-split index

    for (int i = tid; i < M_TOK * 16; i += 256) xs[i >> 4][i & 15] = xa[i];

    v8f acc00 = {}, acc01 = {}, acc10 = {}, acc11 = {};

    // W-tile mapping: thread t -> row t/2 (0..127), K-half (t&1)*64.
    // A 64-wide K slice aligned to 64 == exactly one scale block.
    const int row_w = tid >> 1;
    const int kqw   = (tid & 1) * 64;
    const int o_global   = ntile + row_w;
    const int packed_row = o_global * (IN_F / 2); // int32 index of row start
    // x-tile mapping: thread t -> row t/4 (0..63), K-quarter (t&3)*32
    const int row_x = tid >> 2;
    const int kqx   = (tid & 3) * 32;
    const int kstart = kh * KCHUNK;
    const int k_end  = (kh + 1) * KCHUNK;

    // ---- register staging for the software pipeline ----
    int4  w[8];
    float sc;
#if !HAVE_ASYNC_LDS
    v8h   xr[4];
#endif

    // prologue: fetch stage 0
    {
        const int base = packed_row + ((kstart + kqw) >> 1);
        #pragma unroll
        for (int i = 0; i < 8; ++i) w[i] = *(const int4*)(packed + base + i * 4);
        sc = scales[o_global * (IN_F / 64) + ((kstart + kqw) >> 6)];
#if !HAVE_ASYNC_LDS
        const _Float16* src = x16 + (size_t)row_x * IN_F + kstart + kqx;
        #pragma unroll
        for (int i = 0; i < 4; ++i) xr[i] = *(const v8h*)(src + i * 8);
#endif
    }

    for (int k0 = kstart; k0 < k_end; k0 += KT) {
        __syncthreads();   // previous stage's compute done before overwrite

#if HAVE_ASYNC_LDS
        // ---- x tile: DMA straight into LDS, no VGPR round trip ----
        {
            const _Float16* src = x16 + (size_t)row_x * IN_F + k0 + kqx;
            _Float16* dst = &tA[row_x][kqx];
            #pragma unroll
            for (int j = 0; j < 4; ++j)
                __builtin_amdgcn_global_load_async_to_lds_b128(
                    (av1_v4i*)(src + j * 8),
                    (av3_v4i*)(dst + j * 8),
                    0, 0);
        }
#else
        #pragma unroll
        for (int j = 0; j < 4; ++j) *(v8h*)&tA[row_x][kqx + j * 8] = xr[j];
#endif

        // ---- W tile: packed-magic dequant -> LDS (8 x ds_store_b128) ----
        {
            _Float16 sh = (_Float16)sc;
            v2h s2 = { sh, sh };
            #pragma unroll
            for (int j = 0; j < 8; ++j) {
                v2h p0 = dq2(w[j].x, s2);
                v2h p1 = dq2(w[j].y, s2);
                v2h p2 = dq2(w[j].z, s2);
                v2h p3 = dq2(w[j].w, s2);
                v4h q01 = __builtin_shufflevector(p0, p1, 0, 1, 2, 3);
                v4h q23 = __builtin_shufflevector(p2, p3, 0, 1, 2, 3);
                *(v8h*)&tB[row_w][kqw + j * 8] =
                    __builtin_shufflevector(q01, q23, 0, 1, 2, 3, 4, 5, 6, 7);
            }
        }

        // ---- issue NEXT stage's global loads; they overlap the WMMAs ----
        if (k0 + KT < k_end) {
            const int base = packed_row + ((k0 + KT + kqw) >> 1);
            #pragma unroll
            for (int i = 0; i < 8; ++i) w[i] = *(const int4*)(packed + base + i * 4);
            sc = scales[o_global * (IN_F / 64) + ((k0 + KT + kqw) >> 6)];
#if !HAVE_ASYNC_LDS
            const _Float16* src = x16 + (size_t)row_x * IN_F + k0 + KT + kqx;
            #pragma unroll
            for (int i = 0; i < 4; ++i) xr[i] = *(const v8h*)(src + i * 8);
#endif
        }

#if HAVE_ASYNC_LDS
        wait_async0();     // async x-tile landed in LDS
#endif
        __syncthreads();

        // ---- 16 WMMAs over 4 chunks of K=32, 2x2 register tile ----
        #pragma unroll
        for (int kk = 0; kk < KT; kk += 32) {
            v16h a0 = ld_frag(&tA[m0 + nl][0],      kk, lane);
            v16h a1 = ld_frag(&tA[m0 + 16 + nl][0], kk, lane);
            v16h b0 = ld_frag(&tB[n0 + nl][0],      kk, lane);
            v16h b1 = ld_frag(&tB[n0 + 16 + nl][0], kk, lane);
            acc00 = __builtin_amdgcn_wmma_f32_16x16x32_f16(
                        false, a0, false, b0, (short)0, acc00, false, false);
            acc01 = __builtin_amdgcn_wmma_f32_16x16x32_f16(
                        false, a0, false, b1, (short)0, acc01, false, false);
            acc10 = __builtin_amdgcn_wmma_f32_16x16x32_f16(
                        false, a1, false, b0, (short)0, acc10, false, false);
            acc11 = __builtin_amdgcn_wmma_f32_16x16x32_f16(
                        false, a1, false, b1, (short)0, acc11, false, false);
        }
    }

    // ---- Epilogue: C/D layout -> partial buffer; fold LoRA into split 0 ----
    const int mhi = (lane >> 4) << 3;
    float* pout = part + (size_t)kh * (M_TOK * OUT_F);
    const bool add_lora = (kh == 0);

    #pragma unroll
    for (int ni = 0; ni < 2; ++ni) {
        const int col = ntile + n0 + ni * 16 + nl;
        const v8f aT = ni ? acc01 : acc00;   // top 16 rows (m0..m0+15)
        const v8f aB = ni ? acc11 : acc10;   // bottom 16 rows

        float lb[16];
        if (add_lora) {
            #pragma unroll
            for (int r = 0; r < 16; ++r) lb[r] = loraB[col * 16 + r];
        }

        #pragma unroll
        for (int vv = 0; vv < 8; ++vv) {
            int m  = m0 + vv + mhi;
            int m1 = m + 16;
            float v0 = aT[vv];
            float v1 = aB[vv];
            if (add_lora) {
                float d0 = 0.f, d1 = 0.f;
                #pragma unroll
                for (int r = 0; r < 16; ++r) { d0 += xs[m][r] * lb[r]; d1 += xs[m1][r] * lb[r]; }
                v0 += 2.0f * d0;          // SCALING = alpha/r = 2
                v1 += 2.0f * d1;
            }
            pout[(size_t)m  * OUT_F + col] = v0;
            pout[(size_t)m1 * OUT_F + col] = v1;
        }
    }
}

// ---------------------------------------------------------------------------
// Sum the KSPLIT partials -> d_out (deterministic, no float atomics)
// ---------------------------------------------------------------------------
__global__ void __launch_bounds__(256) combine_kernel(const float* __restrict__ part,
                                                      float* __restrict__ out) {
    const size_t N = (size_t)M_TOK * OUT_F;
    size_t i = ((size_t)blockIdx.x * 256 + threadIdx.x) * 4;
    float4 a = *(const float4*)(part + i);
    float4 b = *(const float4*)(part + N + i);
    float4 c = *(const float4*)(part + 2 * N + i);
    float4 d = *(const float4*)(part + 3 * N + i);
    float4 r;
    r.x = a.x + b.x + c.x + d.x;
    r.y = a.y + b.y + c.y + d.y;
    r.z = a.z + b.z + c.z + d.z;
    r.w = a.w + b.w + c.w + d.w;
    *(float4*)(out + i) = r;
}

extern "C" void kernel_launch(void* const* d_in, const int* in_sizes, int n_in,
                              void* d_out, int out_size, void* d_ws, size_t ws_size,
                              hipStream_t stream) {
    (void)in_sizes; (void)n_in; (void)out_size; (void)ws_size;

    const float* x      = (const float*)d_in[0];   // [1,64,8192] f32
    const int*   packed = (const int*)  d_in[1];   // [NUMEL/2] int32 (byte each)
    const float* scales = (const float*)d_in[2];   // [NUMEL/64] f32
    const float* loraA  = (const float*)d_in[3];   // [16,8192] f32
    const float* loraB  = (const float*)d_in[4];   // [8192,16] f32
    float*       out    = (float*)d_out;           // [1,64,8192] f32

    // workspace layout
    _Float16* x16 = (_Float16*)d_ws;                                    // 1 MB
    float*    xa  = (float*)((char*)d_ws + (1u << 20));                 // 4 KB
    float*    prt = (float*)((char*)d_ws + (1u << 20) + (1u << 16));    // 8 MB

    // 1) x -> f16
    cvt_kernel<<<(M_TOK * IN_F) / 1024, 256, 0, stream>>>(x, x16);
    // 2) LoRA projection xa = x @ A^T
    xa_kernel<<<M_TOK * 16, 256, 0, stream>>>(x, loraA, xa);
    // 3) main WMMA GEMM with 4-bit dequant, split-K partials
    dim3 g(OUT_F / NT, KSPLIT);
    qlora_gemm_kernel<<<g, 256, 0, stream>>>(packed, scales, loraB, x16, xa, prt);
    // 4) reduce partials
    combine_kernel<<<(M_TOK * OUT_F) / 1024, 256, 0, stream>>>(prt, out);
}
